// TransformerBlock_56968446214397
// MI455X (gfx1250) — matrix operand order repaired and checked by reading, not compiled, not run on hardware
//
#include <hip/hip_runtime.h>
#include <stdint.h>

#define B_  4
#define L_  2048
#define DQ_ 256
#define H_  8
#define D_  256
#define HD_ (H_*D_)     // 2048
#define ML_ (B_*L_)     // 8192

static constexpr float NEGI = -4294967295.0f;          // -(2^32 - 1)
static constexpr float INV_TEMP = 1.0f / 16.0000001f;  // 1/(sqrt(256)+1e-6)

typedef __attribute__((ext_vector_type(16))) _Float16 v16h;
typedef __attribute__((ext_vector_type(8)))  float    v8f;
typedef __attribute__((ext_vector_type(4)))  unsigned v4u;
typedef __attribute__((ext_vector_type(4)))  int      v4i;

// WMMA f16 fragment: A and B = one 16-half run each -> 2 x b128 LDS loads
union ABfrag { v16h v; v4u q[2]; unsigned u[8]; };

static __device__ inline unsigned short h2u(_Float16 h) {
    return __builtin_bit_cast(unsigned short, h);
}
static __device__ inline unsigned pack2(float a, float b) {
    return (unsigned)h2u((_Float16)a) | ((unsigned)h2u((_Float16)b) << 16);
}

static __device__ inline v8f wmma16(const v16h& a, const v16h& b, v8f c) {
    // D = A(16x32 f16) * B(32x16 f16) + C(16x16 f32)
    return __builtin_amdgcn_wmma_f32_16x16x32_f16(false, a, false, b, (short)0, c,
                                                  false, false);
}

// lane-xor permute within 16-lane groups, pure VALU (v_permlane16_b32)
static __device__ inline float permx(float v, unsigned s0, unsigned s1) {
    unsigned u = __builtin_bit_cast(unsigned, v);
    unsigned r = __builtin_amdgcn_permlane16(u, u, s0, s1, false, false);
    return __builtin_bit_cast(float, r);
}
// max across the 16 lanes holding one C-layout row (halves stay separate)
static __device__ inline float xmax16(float v) {
    v = fmaxf(v, permx(v, 0x67452301u, 0xEFCDAB89u)); // xor 1
    v = fmaxf(v, permx(v, 0x54761032u, 0xDCFE98BAu)); // xor 2
    v = fmaxf(v, permx(v, 0x32107654u, 0xBA98FEDCu)); // xor 4
    v = fmaxf(v, permx(v, 0xFEDCBA98u, 0x76543210u)); // xor 8
    return v;
}

// CDNA5 async global->LDS copy (ASYNCcnt), with synchronous fallback.
// Builtin signature (from hipcc diagnostics): param0 = v4i in AS(1) (__device__),
// param1 = LDS destination, then imm offset + cpol.
#if __has_builtin(__builtin_amdgcn_global_load_async_to_lds_b128) && \
    __has_builtin(__builtin_amdgcn_s_wait_asynccnt)
#define HAVE_ASYNC_LDS 1
typedef __attribute__((address_space(1))) v4i* as1_v4i;
typedef __attribute__((address_space(3))) v4i* as3_v4i;
#define GASYNC_B128(g, l)                                                   \
    __builtin_amdgcn_global_load_async_to_lds_b128(                         \
        (as1_v4i)(g), (as3_v4i)(l), 0, 0)
#else
#define HAVE_ASYNC_LDS 0
#define GASYNC_B128(g, l) (*(v4u*)(l) = *(const v4u*)(g))
#endif

// ---------------------------------------------------------------------------
// Generic WMMA GEMM: C[M,N] = op(escale * (A[M,K] * B[K,N]) (+bias) (relu))
// Block: 256 threads = 8 waves, 4(M) x 2(N) wave grid, each wave 16x32 out.
// Block tile 64(M) x 64(N), K-step 32.
// LAYOUT: 0 = row-major [M,N]; 1 = per-head [b][h][l][d]; 2 = per-head
// transposed [b][h][d][l] (used for V so flash attention never transposes).
// ---------------------------------------------------------------------------
template <bool A_HALF, bool B_TRANS, bool OUT_HALF, int LAYOUT, bool RELU, bool BIAS>
__global__ __launch_bounds__(256)
void gemm_wmma(const void* __restrict__ Ap, const float* __restrict__ Bp,
               const float* __restrict__ bias, void* __restrict__ Cp,
               int M, int N, int K, float escale)
{
    __shared__ alignas(16) _Float16 Abuf[64][40];   // [m][k], rows 80B (16B mult)
    __shared__ alignas(16) _Float16 Bbuf[64][40];   // [n][k] (B stored transposed)

    const int t    = threadIdx.x;
    const int wid  = t >> 5;
    const int lane = t & 31;
    const int half = lane >> 4;
    const int ln   = lane & 15;
    const int m0   = blockIdx.y * 64;
    const int n0   = blockIdx.x * 64;
    const int wm   = (wid >> 1) * 16;   // 0,16,32,48
    const int wn   = (wid & 1) * 32;    // 0,32

    const v8f vzero = {0.f, 0.f, 0.f, 0.f, 0.f, 0.f, 0.f, 0.f};
    v8f acc0 = vzero, acc1 = vzero;

    for (int kt = 0; kt < K; kt += 32) {
        __syncthreads();
        // --- A tile: 64 rows x 32 k, one b128 per thread ---
        {
            const int row = t >> 2;           // 0..63
            const int kc  = (t & 3) * 8;      // 0,8,16,24
            if constexpr (A_HALF) {
                const _Float16* Ah = (const _Float16*)Ap;
                v4u vv = *(const v4u*)(Ah + (size_t)(m0 + row) * K + kt + kc);
                *(v4u*)&Abuf[row][kc] = vv;
            } else {
                const float* Af = (const float*)Ap;
                const float4* s = (const float4*)(Af + (size_t)(m0 + row) * K + kt + kc);
                float4 x0 = s[0], x1 = s[1];
                v4u pk = {pack2(x0.x, x0.y), pack2(x0.z, x0.w),
                          pack2(x1.x, x1.y), pack2(x1.z, x1.w)};
                *(v4u*)&Abuf[row][kc] = pk;
            }
        }
        // --- B tile into Bbuf[n][k]: 64 n x 32 k ---
        if constexpr (B_TRANS) {       // element (k,n) = W[n*K + k] : contiguous in k
            const int n  = t >> 2;     // 0..63
            const int kc = (t & 3) * 8;
            const float* s = Bp + (size_t)(n0 + n) * K + kt + kc;
            float4 x0 = *(const float4*)s, x1 = *(const float4*)(s + 4);
            v4u pk = {pack2(x0.x, x0.y), pack2(x0.z, x0.w),
                      pack2(x1.x, x1.y), pack2(x1.z, x1.w)};
            *(v4u*)&Bbuf[n][kc] = pk;
        } else {                       // element (k,n) = W[k*N + n] : contiguous in n
            const int k  = t >> 3;     // 0..31
            const int nc = (t & 7) * 8;
            const float* s = Bp + (size_t)(kt + k) * N + n0 + nc;
            float4 x0 = *(const float4*)s, x1 = *(const float4*)(s + 4);
            float vs[8] = {x0.x, x0.y, x0.z, x0.w, x1.x, x1.y, x1.z, x1.w};
            #pragma unroll
            for (int i = 0; i < 8; ++i) Bbuf[nc + i][k] = (_Float16)vs[i];
        }
        __syncthreads();

        // --- fragments: 2 x b128 each; one A-frag feeds two WMMAs ---
        ABfrag a, b0, b1;
        const char* arow  = (const char*)&Abuf[wm + ln][0];
        const char* brow0 = (const char*)&Bbuf[wn + ln][0];
        const char* brow1 = (const char*)&Bbuf[wn + 16 + ln][0];
        a.q[0]  = *(const v4u*)(arow + 32 * half);
        a.q[1]  = *(const v4u*)(arow + 32 * half + 16);
        b0.q[0] = *(const v4u*)(brow0 + 32 * half);
        b0.q[1] = *(const v4u*)(brow0 + 32 * half + 16);
        b1.q[0] = *(const v4u*)(brow1 + 32 * half);
        b1.q[1] = *(const v4u*)(brow1 + 32 * half + 16);
        acc0 = wmma16(a.v, b0.v, acc0);
        acc1 = wmma16(a.v, b1.v, acc1);
    }

    // --- epilogue: C layout is VGPR r -> row r+8*half, lane%16 -> col ---
    #pragma unroll
    for (int tile = 0; tile < 2; ++tile) {
        const v8f acc = tile ? acc1 : acc0;
        #pragma unroll
        for (int r = 0; r < 8; ++r) {
            const int row = m0 + wm + r + 8 * half;
            const int col = n0 + wn + tile * 16 + ln;
            float v = acc[r] * escale;
            if constexpr (BIAS) v += bias[col];
            if constexpr (RELU) v = fmaxf(v, 0.f);
            size_t idx;
            if constexpr (LAYOUT == 1) {        // -> T[b][h][l][d]
                const int bb = row >> 11, l = row & 2047;
                const int hh = col >> 8,  dd = col & 255;
                idx = (((size_t)(bb * H_ + hh) * L_) + l) * D_ + dd;
            } else if constexpr (LAYOUT == 2) { // -> T[b][h][d][l]
                const int bb = row >> 11, l = row & 2047;
                const int hh = col >> 8,  dd = col & 255;
                idx = (((size_t)(bb * H_ + hh) * D_) + dd) * L_ + l;
            } else {
                idx = (size_t)row * N + col;
            }
            if constexpr (OUT_HALF) ((_Float16*)Cp)[idx] = (_Float16)v;
            else                    ((float*)Cp)[idx]    = v;
        }
    }
}

// ---------------------------------------------------------------------------
// Flash attention: block = (b, h, 128 query rows), 8 waves, each wave owns a
// 16(q) x 256(d) output. K tiles key-major, V tiles d-major (V is stored
// pre-transposed per head in global, so staging is pure contiguous copy).
// Double-buffered async global->LDS staging overlaps fills with WMMA compute.
// Row max via v_permlane16, row sums via P x ones WMMA. Qh pre-scaled 1/TEMP.
// ---------------------------------------------------------------------------
__global__ __launch_bounds__(256)
void flash_attn(const _Float16* __restrict__ Qh, const _Float16* __restrict__ Kh,
                const _Float16* __restrict__ VhT, const unsigned char* __restrict__ padmask,
                _Float16* __restrict__ Vatt)
{
    __shared__ alignas(16) _Float16 Kt[2][32][D_ + 8]; // [buf][key][d], row 528B
    __shared__ alignas(16) _Float16 Vt[2][D_][40];     // [buf][d][key], row 80B
    __shared__ alignas(16) _Float16 Pbuf[8][16][32];   // per-wave P re-striping

    const int t    = threadIdx.x;
    const int wid  = t >> 5;
    const int lane = t & 31;
    const int half = lane >> 4;
    const int ln   = lane & 15;
    const int b    = blockIdx.z;
    const int h    = blockIdx.y;
    const int qbase = blockIdx.x * 128;
    const int q0    = qbase + wid * 16;
    const size_t headoff = (size_t)(b * H_ + h) * L_ * D_;  // same for Kh and VhT

    // staging geometry (per thread): K: one (key, 32-d slab); V: one d-row
    const int skey = t & 31;
    const int sdc  = (t >> 5) * 32;
    const _Float16* Krow = Kh  + headoff + (size_t)skey * D_ + sdc;   // key = kb+skey
    const _Float16* Vrow = VhT + headoff + (size_t)t * L_;            // row d=t, col = key

    // issue one tile's fill (async if available, else synchronous copy)
    auto stage = [&](int kb, int buf) {
        const _Float16* kg = Krow + (size_t)kb * D_;
        #pragma unroll
        for (int i = 0; i < 4; ++i)
            GASYNC_B128(kg + 8 * i, &Kt[buf][skey][sdc + 8 * i]);
        const _Float16* vg = Vrow + kb;
        #pragma unroll
        for (int i = 0; i < 4; ++i)
            GASYNC_B128(vg + 8 * i, &Vt[buf][t][8 * i]);
    };

    // Q fragments for all 8 d-chunks (A layout): 2 x global b128 per chunk
    ABfrag aQ[8];
    {
        const char* Qrow = (const char*)(Qh + headoff + (size_t)(q0 + ln) * D_);
        #pragma unroll
        for (int c = 0; c < 8; ++c) {
            aQ[c].q[0] = *(const v4u*)(Qrow + 64 * c + 32 * half);
            aQ[c].q[1] = *(const v4u*)(Qrow + 64 * c + 32 * half + 16);
        }
    }

    ABfrag bOne;                            // all-1.0 f16 B fragment (row-sum trick)
    #pragma unroll
    for (int j = 0; j < 8; ++j) bOne.u[j] = 0x3C003C00u;

    const v8f vzero = {0.f, 0.f, 0.f, 0.f, 0.f, 0.f, 0.f, 0.f};
    v8f acc[16];
    #pragma unroll
    for (int i = 0; i < 16; ++i) acc[i] = vzero;

    float mrow[8], lrow[8];
    bool  rowm[8];
    bool  anyr = false;
    #pragma unroll
    for (int r = 0; r < 8; ++r) {
        mrow[r] = -1e30f;
        lrow[r] = 0.f;
        rowm[r] = padmask[b * L_ + q0 + r + 8 * half] != 0;
        anyr = anyr || rowm[r];
    }
    const bool anym = (__ballot(anyr) != 0ull);   // wave-uniform

    const int nsteps = (qbase + 128) / 32;        // block needs keys < qbase+128
    stage(0, 0);                                  // prologue fill of buffer 0

    for (int i = 0; i < nsteps; ++i) {
        const int kb  = i * 32;
        const int cur = i & 1;
#if HAVE_ASYNC_LDS
        __builtin_amdgcn_s_wait_asynccnt(0);      // this wave's fills landed
#endif
        __syncthreads();                          // tile[cur] visible; tile[1-cur] free
        if (i + 1 < nsteps) stage(kb + 32, 1 - cur);

        // wave-uniform causal skip: this wave's rows end at q0+15
        if (kb > q0 + 15) continue;

        // scores S = Q * K^T for 32 keys: two 16x16 tiles, 8 WMMAs each
        v8f s0 = vzero, s1 = vzero;
        #pragma unroll
        for (int c = 0; c < 8; ++c) {
            ABfrag bK0, bK1;
            const char* kr0 = (const char*)&Kt[cur][ln][0]      + 64 * c + 32 * half;
            const char* kr1 = (const char*)&Kt[cur][ln + 16][0] + 64 * c + 32 * half;
            bK0.q[0] = *(const v4u*)(kr0);
            bK0.q[1] = *(const v4u*)(kr0 + 16);
            bK1.q[0] = *(const v4u*)(kr1);
            bK1.q[1] = *(const v4u*)(kr1 + 16);
            s0 = wmma16(aQ[c].v, bK0.v, s0);
            s1 = wmma16(aQ[c].v, bK1.v, s1);
        }

        // online softmax per row; masking only on diagonal tiles / padded rows
        const bool need_mask = anym || (kb + 31 > q0);   // wave-uniform
        float sar[8];
        #pragma unroll
        for (int r = 0; r < 8; ++r) {
            const int q = q0 + r + 8 * half;
            float v0 = s0[r];
            float v1 = s1[r];
            if (need_mask) {
                if (kb + ln      > q || rowm[r]) v0 = NEGI;
                if (kb + 16 + ln > q || rowm[r]) v1 = NEGI;
            }
            const float mx   = xmax16(fmaxf(v0, v1));    // pure-VALU 16-lane max
            const float mnew = fmaxf(mrow[r], mx);
            const float scale = __expf(mrow[r] - mnew);
            const float p0 = __expf(v0 - mnew);
            const float p1 = __expf(v1 - mnew);
            sar[r]  = scale;
            mrow[r] = mnew;
            #pragma unroll
            for (int tt = 0; tt < 16; ++tt) acc[tt][r] *= scale;
            Pbuf[wid][r + 8 * half][ln]      = (_Float16)p0;
            Pbuf[wid][r + 8 * half][ln + 16] = (_Float16)p1;
        }

        // restripe P (C layout -> A layout) via wave-private LDS: 2 x b128
        ABfrag aP;
        {
            const char* pr = (const char*)&Pbuf[wid][ln][0];
            aP.q[0] = *(const v4u*)(pr + 32 * half);
            aP.q[1] = *(const v4u*)(pr + 32 * half + 16);
        }

        // row sums of (f16-rounded) P via the matrix pipe: P x ones
        {
            v8f rsv = wmma16(aP.v, bOne.v, vzero);
            #pragma unroll
            for (int r = 0; r < 8; ++r) lrow[r] = lrow[r] * sar[r] + rsv[r];
        }

        // O += P * V : 16 d-chunks, B-frag = 2 x b128 from d-major Vt
        #pragma unroll
        for (int tt = 0; tt < 16; tt += 2) {
            ABfrag bV0, bV1;
            const char* vr0 = (const char*)&Vt[cur][tt * 16 + ln][0]       + 32 * half;
            const char* vr1 = (const char*)&Vt[cur][(tt + 1) * 16 + ln][0] + 32 * half;
            bV0.q[0] = *(const v4u*)(vr0);
            bV0.q[1] = *(const v4u*)(vr0 + 16);
            bV1.q[0] = *(const v4u*)(vr1);
            bV1.q[1] = *(const v4u*)(vr1 + 16);
            acc[tt]     = wmma16(aP.v, bV0.v, acc[tt]);
            acc[tt + 1] = wmma16(aP.v, bV1.v, acc[tt + 1]);
        }
    }
#if HAVE_ASYNC_LDS
    __builtin_amdgcn_s_wait_asynccnt(0);          // drain before wave exit
#endif

    // normalize and store V_att in (B, L, H*D) f16 layout
    #pragma unroll
    for (int r = 0; r < 8; ++r) {
        const int q = q0 + r + 8 * half;
        const float inv = 1.0f / lrow[r];
        const size_t base = ((size_t)(b * L_) + q) * HD_ + h * D_;
        #pragma unroll
        for (int tt = 0; tt < 16; ++tt)
            Vatt[base + tt * 16 + ln] = (_Float16)(acc[tt][r] * inv);
    }
}

// ---------------------------------------------------------------------------
// y = LayerNorm(A_row + B_row) * g + beta ; optional f16 mirror for next GEMM
// One block (256 threads) per row of 256 elements.
// ---------------------------------------------------------------------------
__global__ __launch_bounds__(256)
void add_layernorm(const float* __restrict__ A, const float* __restrict__ Bv,
                   const float* __restrict__ g, const float* __restrict__ beta,
                   float* __restrict__ Of, _Float16* __restrict__ Obf)
{
    __shared__ float red[8];
    const int row = blockIdx.x, t = threadIdx.x;
    const size_t idx = (size_t)row * 256 + t;
    const float x = A[idx] + Bv[idx];

    float s = x;
    #pragma unroll
    for (int off = 16; off >= 1; off >>= 1) s += __shfl_xor(s, off, 32);
    if ((t & 31) == 0) red[t >> 5] = s;
    __syncthreads();
    const float mu = (red[0] + red[1] + red[2] + red[3] +
                      red[4] + red[5] + red[6] + red[7]) * (1.0f / 256.0f);
    __syncthreads();

    const float d = x - mu;
    float s2 = d * d;
    #pragma unroll
    for (int off = 16; off >= 1; off >>= 1) s2 += __shfl_xor(s2, off, 32);
    if ((t & 31) == 0) red[t >> 5] = s2;
    __syncthreads();
    const float var = (red[0] + red[1] + red[2] + red[3] +
                       red[4] + red[5] + red[6] + red[7]) * (1.0f / 256.0f);

    const float y = g[t] * d * rsqrtf(var + 1e-5f) + beta[t];
    Of[idx] = y;
    if (Obf) Obf[idx] = (_Float16)y;
}

// ---------------------------------------------------------------------------
extern "C" void kernel_launch(void* const* d_in, const int* in_sizes, int n_in,
                              void* d_out, int out_size, void* d_ws, size_t ws_size,
                              hipStream_t stream)
{
    (void)in_sizes; (void)n_in; (void)out_size; (void)ws_size;
    const float* Q      = (const float*)d_in[0];
    const float* K      = (const float*)d_in[1];
    const float* V      = (const float*)d_in[2];
    const unsigned char* mask = (const unsigned char*)d_in[3];
    const float* W_q    = (const float*)d_in[4];
    const float* W_k    = (const float*)d_in[5];
    const float* W_v    = (const float*)d_in[6];
    const float* W_o    = (const float*)d_in[7];
    const float* lin1_w = (const float*)d_in[8];
    const float* lin1_b = (const float*)d_in[9];
    const float* lin2_w = (const float*)d_in[10];
    const float* lin2_b = (const float*)d_in[11];
    const float* ln_g   = (const float*)d_in[12];
    const float* ln_b   = (const float*)d_in[13];

    char* w = (char*)d_ws;
    // f16 head tensors: B*H*L*D * 2B = 32 MiB each
    _Float16* Qh   = (_Float16*)(w + 0);
    _Float16* Kh   = (_Float16*)(w + 33554432u);
    _Float16* VhT  = (_Float16*)(w + 67108864u);   // [b][h][d][l]
    _Float16* Vatt = (_Float16*)(w + 100663296u);
    // post-attention intermediates reuse the (dead) Qh region
    float*    attn = (float*)(w + 0);            // 8 MiB f32
    float*    X    = (float*)(w + 8388608u);     // 8 MiB f32
    _Float16* Xbf  = (_Float16*)(w + 16777216u); // 4 MiB f16
    _Float16* H1   = (_Float16*)(w + 20971520u); // 4 MiB f16
    float*    F    = (float*)(w + 25165824u);    // 8 MiB f32

    const dim3 blk(256);

    // 1) QKV projections: [8192 x 256] x [256 x 2048] -> f16 head tensors.
    //    Qh pre-scaled by 1/TEMP; V written transposed per head.
    const dim3 g1(HD_ / 64, ML_ / 64);
    gemm_wmma<false, false, true, 1, false, false><<<g1, blk, 0, stream>>>(Q, W_q, nullptr, Qh, ML_, HD_, DQ_, INV_TEMP);
    gemm_wmma<false, false, true, 1, false, false><<<g1, blk, 0, stream>>>(K, W_k, nullptr, Kh, ML_, HD_, DQ_, 1.0f);
    gemm_wmma<false, false, true, 2, false, false><<<g1, blk, 0, stream>>>(V, W_v, nullptr, VhT, ML_, HD_, DQ_, 1.0f);

    // 2) flash attention -> Vatt (B, L, H*D) f16
    const dim3 g2(L_ / 128, H_, B_);
    flash_attn<<<g2, blk, 0, stream>>>(Qh, Kh, VhT, mask, Vatt);

    // 3) output projection: [8192 x 2048] x [2048 x 256] -> f32
    const dim3 g3(D_ / 64, ML_ / 64);
    gemm_wmma<true, false, false, 0, false, false><<<g3, blk, 0, stream>>>(Vatt, W_o, nullptr, attn, ML_, D_, HD_, 1.0f);

    // 4) X = LN(Q + attn) ; keep f32 + f16 copies
    add_layernorm<<<dim3(ML_), blk, 0, stream>>>(Q, attn, ln_g, ln_b, X, Xbf);

    // 5) H1 = relu(X @ lin1_w^T + b1) -> f16
    gemm_wmma<true, true, true, 0, true, true><<<g3, blk, 0, stream>>>(Xbf, lin1_w, lin1_b, H1, ML_, D_, D_, 1.0f);

    // 6) F = H1 @ lin2_w^T + b2 -> f32
    gemm_wmma<true, true, false, 0, false, true><<<g3, blk, 0, stream>>>(H1, lin2_w, lin2_b, F, ML_, D_, D_, 1.0f);

    // 7) out = LN(F + X)
    add_layernorm<<<dim3(ML_), blk, 0, stream>>>(F, X, ln_g, ln_b, (float*)d_out, nullptr);
}
